// Attention_77799037600056
// MI455X (gfx1250) — matrix operand order repaired and checked by reading, not compiled, not run on hardware
//
#include <hip/hip_runtime.h>

// ---------------------------------------------------------------------------
// Bahdanau attention for MI455X (gfx1250, wave32, WMMA + TDM).
// S=1024, B=32, F=1024, D=512. Dominant GEMM: enc @ W1enc^T (34.4 GFLOP)
// -> v_wmma_f32_16x16x32_bf16, A tiles DMA'd into LDS by the Tensor Data
// Mover (tensor_load_to_lds + s_wait_tensorcnt), double buffered.
// ---------------------------------------------------------------------------

#define S_LEN  1024
#define BATCH  32
#define ENC_F  1024
#define DHID   512
#define IN_DIM 1536

typedef __attribute__((ext_vector_type(16))) __bf16   bf16x16;
typedef __attribute__((ext_vector_type(8)))  float    f32x8;
typedef __attribute__((ext_vector_type(4)))  float    f32x4;
typedef __attribute__((ext_vector_type(4)))  unsigned u32x4;
typedef __attribute__((ext_vector_type(8)))  int      i32x8;
typedef __attribute__((ext_vector_type(4)))  int      i32x4;

union FragBF { bf16x16 f; u32x4 q[2]; };   // 16 bf16 per lane = 2 x b128

// round-to-nearest-even f32 -> bf16, packed pairwise
__device__ __forceinline__ unsigned pack_bf16x2(float a, float b) {
  unsigned ua = __float_as_uint(a);
  unsigned ub = __float_as_uint(b);
  ua = (ua + 0x7FFFu + ((ua >> 16) & 1u)) >> 16;
  ub = (ub + 0x7FFFu + ((ub >> 16) & 1u)) >> 16;
  return (ua & 0xFFFFu) | (ub << 16);
}

// ---------------------------------------------------------------------------
// TDM: issue a 64-row x 256-elem bf16 2D tile load into LDS.
// D# per CDNA5 ISA:
//   group0: count=1, lds_addr, global_addr, type=2
//   group1: data_size=2B, pad_enable (interval=128 dwords -> code 6,
//           amount=4 dwords -> code 3) => LDS row stride 132 dwords,
//           tensor_dim0=1024, tile_dim0=256, tile_dim1=64,
//           tensor_dim0_stride = BATCH*ENC_F = 32768 elements.
// This toolchain exposes the 6-arg builtin:
//   (u32x4 g0, i32x8 g1, i32x4 g2, i32x4 g3, i32x8 pad, i32 cpol)
// ---------------------------------------------------------------------------
__device__ __forceinline__ void tdm_load_tile(unsigned lds_off,
                                              const unsigned short* gptr) {
  unsigned long long ga = (unsigned long long)(uintptr_t)gptr;
  u32x4 g0 = {
    1u,                                               // count=1, user mode
    lds_off,                                          // lds_addr (bytes)
    (unsigned)ga,                                     // global_addr[31:0]
    (unsigned)((ga >> 32) & 0x1FFFFFFu) | (2u << 30)  // addr[56:32] | type=2
  };
  i32x8 g1 = {
    (int)((1u << 16) | (1u << 20) | (6u << 22) | (3u << 25)), // dsize|pad
    (int)(((unsigned)ENC_F & 0xFFFFu) << 16),  // tensor_dim0 = 1024 (lo16)
    0,                                         // dim0 hi, tensor_dim1 lo
    (int)(16u | (256u << 16)),                 // tensor_dim1 hi (1<<20) | tile_dim0=256
    64,                                        // tile_dim1=64, tile_dim2=0
    (int)(BATCH * ENC_F),                      // tensor_dim0_stride lo32
    0, 0
  };
  i32x4 z4 = {0, 0, 0, 0};
  i32x8 z8 = {0, 0, 0, 0, 0, 0, 0, 0};
  __builtin_amdgcn_tensor_load_to_lds(g0, g1, z4, z4, z8, 0);
}

// ---------------------------------------------------------------------------
// Kernel: W1[:, 0:1024] f32 -> bf16 copy [512][1024].  grid 1024 x 256
// ---------------------------------------------------------------------------
__global__ __launch_bounds__(256)
void convert_w1_kernel(const float* __restrict__ W1, unsigned* __restrict__ w1bf) {
  int u = blockIdx.x * 256 + threadIdx.x;   // pair index, 0..262143
  int o = u >> 9;
  int j = u & 511;
  const float* src = W1 + o * IN_DIM + 2 * j;
  w1bf[u] = pack_bf16x2(src[0], src[1]);
}

// ---------------------------------------------------------------------------
// Kernel: encoder_output f32 [S,B,F] -> bf16 copy (64MB, L2-resident).
// 4194304 u32x4 outputs; grid 16384 x 256.
// ---------------------------------------------------------------------------
__global__ __launch_bounds__(256)
void convert_enc_kernel(const float* __restrict__ enc, u32x4* __restrict__ encbf) {
  int q = blockIdx.x * 256 + threadIdx.x;
  const f32x4* src = (const f32x4*)enc;
  f32x4 a = src[2 * q];
  f32x4 c = src[2 * q + 1];
  u32x4 o = { pack_bf16x2(a.x, a.y), pack_bf16x2(a.z, a.w),
              pack_bf16x2(c.x, c.y), pack_bf16x2(c.z, c.w) };
  encbf[q] = o;
}

// ---------------------------------------------------------------------------
// Kernel: g[b,o] = b1[o] + sum_d dec[b,d] * W1[o, 1024+d]
// ---------------------------------------------------------------------------
__global__ __launch_bounds__(256)
void dec_bias_kernel(const float* __restrict__ dec, const float* __restrict__ W1,
                     const float* __restrict__ b1, float* __restrict__ g) {
  const int b    = blockIdx.x;
  const int wave = threadIdx.x >> 5;
  const int lane = threadIdx.x & 31;
  __shared__ float ldec[DHID];
  for (int i = threadIdx.x; i < DHID; i += 256) ldec[i] = dec[b * DHID + i];
  __syncthreads();
  for (int o = wave; o < DHID; o += 8) {
    const float* wrow = W1 + o * IN_DIM + ENC_F;
    float acc = 0.f;
    for (int d = lane; d < DHID; d += 32) acc += ldec[d] * wrow[d];
    acc += __shfl_xor(acc, 1, 32);
    acc += __shfl_xor(acc, 2, 32);
    acc += __shfl_xor(acc, 4, 32);
    acc += __shfl_xor(acc, 8, 32);
    acc += __shfl_xor(acc, 16, 32);
    if (lane == 0) g[b * DHID + o] = acc + b1[o];
  }
}

// ---------------------------------------------------------------------------
// Shared epilogue: tanh + W2 weighting + cross-lane/cross-wave reduce.
// ---------------------------------------------------------------------------
__device__ __forceinline__ void score_epilogue(
    f32x8 acc[4][4], const float* g, const float* W2, float* scores,
    float (*ldsS)[64], int b, int s0, int tid, int wave, int llo, int lhi,
    int col0) {
  float gv[4], w2v[4];
#pragma unroll
  for (int nt = 0; nt < 4; ++nt) {
    int o = col0 + nt * 16 + llo;
    gv[nt]  = g[b * DHID + o];
    w2v[nt] = W2[o];
  }
  float prt[4][8];
#pragma unroll
  for (int mt = 0; mt < 4; ++mt)
#pragma unroll
    for (int v = 0; v < 8; ++v) prt[mt][v] = 0.f;
#pragma unroll
  for (int mt = 0; mt < 4; ++mt)
#pragma unroll
    for (int nt = 0; nt < 4; ++nt) {
      f32x8 c = acc[mt][nt];
#pragma unroll
      for (int v = 0; v < 8; ++v)
        prt[mt][v] += tanhf(c[v] + gv[nt]) * w2v[nt];
    }
#pragma unroll
  for (int mt = 0; mt < 4; ++mt)
#pragma unroll
    for (int v = 0; v < 8; ++v) {
      float r = prt[mt][v];
      r += __shfl_xor(r, 1, 32);
      r += __shfl_xor(r, 2, 32);
      r += __shfl_xor(r, 4, 32);
      r += __shfl_xor(r, 8, 32);
      prt[mt][v] = r;
    }
  if (llo == 0) {
#pragma unroll
    for (int mt = 0; mt < 4; ++mt)
#pragma unroll
      for (int v = 0; v < 8; ++v)
        ldsS[wave][mt * 16 + lhi * 8 + v] = prt[mt][v];   // C row = v + 8*lhi
  }
  __syncthreads();
  if (tid < 64) {
    float s = 0.f;
#pragma unroll
    for (int w = 0; w < 8; ++w) s += ldsS[w][tid];
    scores[b * S_LEN + s0 + tid] = s;      // b2 omitted: cancels in softmax
  }
}

// ---------------------------------------------------------------------------
// Core kernel (TDM path): A tiles (bf16 enc copy) DMA'd into double-buffered
// LDS by the Tensor Data Mover; B straight from the L2-resident bf16 W1 copy.
// 8 waves; wave w owns N columns [w*64, w*64+64); WG tile = 64 rows.
// ---------------------------------------------------------------------------
__global__ __launch_bounds__(256)
void attn_score_tdm_kernel(const unsigned short* __restrict__ encbf, // [S,B,F] bf16
                           const unsigned* __restrict__ w1bf,        // [512][1024] bf16
                           const float* __restrict__ g,
                           const float* __restrict__ W2,
                           float* __restrict__ scores) {
  __shared__ unsigned ldsA[2][64 * 132];  // 2 x (64 rows x 128 pairs + 4 pad)
  __shared__ float    ldsS[8][64];

  const int b    = blockIdx.y;
  const int s0   = blockIdx.x * 64;
  const int tid  = threadIdx.x;
  const int wave = tid >> 5;
  const int lane = tid & 31;
  const int llo  = lane & 15;
  const int lhi  = lane >> 4;
  const int col0 = wave * 64;

  f32x8 acc[4][4];
  const f32x8 zero8 = {0.f, 0.f, 0.f, 0.f, 0.f, 0.f, 0.f, 0.f};
#pragma unroll
  for (int mt = 0; mt < 4; ++mt)
#pragma unroll
    for (int nt = 0; nt < 4; ++nt) acc[mt][nt] = zero8;

  const u32x4* bsrc = (const u32x4*)w1bf;
  const unsigned short* tile0 = encbf + (size_t)(s0 * BATCH + b) * ENC_F;

  if (wave == 0)
    tdm_load_tile((unsigned)(uintptr_t)&ldsA[0][0], tile0);

  for (int kb = 0; kb < 4; ++kb) {            // K chunks of 256
    const unsigned* bufA = ldsA[kb & 1];
    if (wave == 0) __builtin_amdgcn_s_wait_tensorcnt(0);
    __syncthreads();                          // tile kb visible; prev buf free
    if (wave == 0 && kb < 3)
      tdm_load_tile((unsigned)(uintptr_t)&ldsA[(kb + 1) & 1][0],
                    tile0 + (kb + 1) * 256);

    for (int k = 0; k < 256; k += 32) {       // 8 k-steps per chunk
      FragBF a[4];
#pragma unroll
      for (int mt = 0; mt < 4; ++mt) {
        int row = mt * 16 + llo;
        const u32x4* p = (const u32x4*)&bufA[row * 132 + (k >> 1) + lhi * 4];
        a[mt].q[0] = p[0];     // K = lhi*8 + 0..7
        a[mt].q[1] = p[2];     // K = 16 + lhi*8 + 0..7
      }
      FragBF bb[4];
#pragma unroll
      for (int nt = 0; nt < 4; ++nt) {
        int o = col0 + nt * 16 + llo;
        int base = (o * ENC_F + kb * 256 + k + lhi * 8) >> 3;  // u32x4 units
        bb[nt].q[0] = bsrc[base];
        bb[nt].q[1] = bsrc[base + 2];
      }
#pragma unroll
      for (int mt = 0; mt < 4; ++mt)
#pragma unroll
        for (int nt = 0; nt < 4; ++nt)
          acc[mt][nt] = __builtin_amdgcn_wmma_f32_16x16x32_bf16(
              false, a[mt].f, false, bb[nt].f, (short)0, acc[mt][nt],
              false, false);
    }
  }
  score_epilogue(acc, g, W2, scores, ldsS, b, s0, tid, wave, llo, lhi, col0);
}

// ---------------------------------------------------------------------------
// Fallback core kernel (small workspace): pack f32 enc -> bf16 LDS in-kernel.
// ---------------------------------------------------------------------------
__global__ __launch_bounds__(256)
void attn_score_fallback_kernel(const float* __restrict__ enc,
                                const unsigned* __restrict__ w1bf,
                                const float* __restrict__ g,
                                const float* __restrict__ W2,
                                float* __restrict__ scores) {
  __shared__ unsigned ldsA[64 * 132];
  __shared__ float    ldsS[8][64];

  const int b    = blockIdx.y;
  const int s0   = blockIdx.x * 64;
  const int tid  = threadIdx.x;
  const int wave = tid >> 5;
  const int lane = tid & 31;
  const int llo  = lane & 15;
  const int lhi  = lane >> 4;
  const int col0 = wave * 64;

  f32x8 acc[4][4];
  const f32x8 zero8 = {0.f, 0.f, 0.f, 0.f, 0.f, 0.f, 0.f, 0.f};
#pragma unroll
  for (int mt = 0; mt < 4; ++mt)
#pragma unroll
    for (int nt = 0; nt < 4; ++nt) acc[mt][nt] = zero8;

  const f32x4* esrc = (const f32x4*)enc;
  const u32x4* bsrc = (const u32x4*)w1bf;

  for (int kb = 0; kb < 4; ++kb) {
    __syncthreads();
#pragma unroll
    for (int i = 0; i < 16; ++i) {
      int idx = tid + i * 256;
      int row = idx >> 6;
      int q4  = idx & 63;
      f32x4 v = esrc[((s0 + row) * BATCH + b) * (ENC_F / 4) + kb * 64 + q4];
      int o = row * 132 + q4 * 2;
      ldsA[o]     = pack_bf16x2(v.x, v.y);
      ldsA[o + 1] = pack_bf16x2(v.z, v.w);
    }
    __syncthreads();

    for (int k = 0; k < 256; k += 32) {
      FragBF a[4];
#pragma unroll
      for (int mt = 0; mt < 4; ++mt) {
        int row = mt * 16 + llo;
        const u32x4* p = (const u32x4*)&ldsA[row * 132 + (k >> 1) + lhi * 4];
        a[mt].q[0] = p[0];
        a[mt].q[1] = p[2];
      }
      FragBF bb[4];
#pragma unroll
      for (int nt = 0; nt < 4; ++nt) {
        int o = col0 + nt * 16 + llo;
        int base = (o * ENC_F + kb * 256 + k + lhi * 8) >> 3;
        bb[nt].q[0] = bsrc[base];
        bb[nt].q[1] = bsrc[base + 2];
      }
#pragma unroll
      for (int mt = 0; mt < 4; ++mt)
#pragma unroll
        for (int nt = 0; nt < 4; ++nt)
          acc[mt][nt] = __builtin_amdgcn_wmma_f32_16x16x32_bf16(
              false, a[mt].f, false, bb[nt].f, (short)0, acc[mt][nt],
              false, false);
    }
  }
  score_epilogue(acc, g, W2, scores, ldsS, b, s0, tid, wave, llo, lhi, col0);
}

// ---------------------------------------------------------------------------
// Softmax over S per batch.
// ---------------------------------------------------------------------------
__global__ __launch_bounds__(256)
void softmax_kernel(const float* __restrict__ scores, float* __restrict__ alpha) {
  const int b = blockIdx.x, t = threadIdx.x;
  __shared__ float red[256];
  float v[4];
  float m = -3.402823466e38f;
#pragma unroll
  for (int i = 0; i < 4; ++i) {
    v[i] = scores[b * S_LEN + t + i * 256];
    m = fmaxf(m, v[i]);
  }
  red[t] = m;
  __syncthreads();
  for (int off = 128; off > 0; off >>= 1) {
    if (t < off) red[t] = fmaxf(red[t], red[t + off]);
    __syncthreads();
  }
  m = red[0];
  __syncthreads();
  float sum = 0.f;
#pragma unroll
  for (int i = 0; i < 4; ++i) {
    v[i] = expf(v[i] - m);
    sum += v[i];
  }
  red[t] = sum;
  __syncthreads();
  for (int off = 128; off > 0; off >>= 1) {
    if (t < off) red[t] += red[t + off];
    __syncthreads();
  }
  float inv = 1.0f / red[0];
#pragma unroll
  for (int i = 0; i < 4; ++i)
    alpha[b * S_LEN + t + i * 256] = v[i] * inv;
}

// ---------------------------------------------------------------------------
// Context from the L2-resident bf16 enc copy: a[b,f] = sum_s alpha*enc.
// Thread owns a bf16 pair; grid (B, 4), 128 threads.
// ---------------------------------------------------------------------------
__global__ __launch_bounds__(128)
void context_bf16_kernel(const unsigned* __restrict__ encbf,
                         const float* __restrict__ alpha,
                         float* __restrict__ out) {
  const int b = blockIdx.x;
  const int p = blockIdx.y * 128 + threadIdx.x;   // pair index within row
  __shared__ float la[S_LEN];
  for (int i = threadIdx.x; i < S_LEN; i += 128) la[i] = alpha[b * S_LEN + i];
  __syncthreads();
  float a0 = 0.f, a1 = 0.f;
#pragma unroll 4
  for (int s = 0; s < S_LEN; ++s) {
    unsigned pr = encbf[(s * BATCH + b) * (ENC_F / 2) + p];
    float w = la[s];
    a0 += w * __uint_as_float(pr << 16);
    a1 += w * __uint_as_float(pr & 0xFFFF0000u);
  }
  out[b * ENC_F + 2 * p]     = a0;
  out[b * ENC_F + 2 * p + 1] = a1;
}

// Fallback context: read f32 enc directly.
__global__ __launch_bounds__(128)
void context_f32_kernel(const float* __restrict__ enc,
                        const float* __restrict__ alpha,
                        float* __restrict__ out) {
  const int b = blockIdx.x;
  const int f = blockIdx.y * 128 + threadIdx.x;
  __shared__ float la[S_LEN];
  for (int i = threadIdx.x; i < S_LEN; i += 128) la[i] = alpha[b * S_LEN + i];
  __syncthreads();
  float accv = 0.f;
#pragma unroll 4
  for (int s = 0; s < S_LEN; ++s)
    accv += la[s] * enc[(s * BATCH + b) * ENC_F + f];
  out[b * ENC_F + f] = accv;
}

// ---------------------------------------------------------------------------
extern "C" void kernel_launch(void* const* d_in, const int* in_sizes, int n_in,
                              void* d_out, int out_size, void* d_ws, size_t ws_size,
                              hipStream_t stream) {
  const float* enc = (const float*)d_in[0];   // [1024,32,1024]
  const float* dec = (const float*)d_in[1];   // [32,512]
  const float* W1  = (const float*)d_in[2];   // [512,1536]
  const float* b1  = (const float*)d_in[3];   // [512]
  const float* W2  = (const float*)d_in[4];   // [1,512]
  // d_in[5] = b2: constant shift over S, cancels in softmax
  float* out = (float*)d_out;                 // [32,1,1024]

  char* ws = (char*)d_ws;
  const size_t off_g     = (size_t)1u << 20;            // after 1MB w1bf
  const size_t off_score = off_g + ((size_t)64u << 10);
  const size_t off_alpha = off_score + ((size_t)128u << 10);
  const size_t off_encbf = (size_t)2u << 20;
  const size_t need_tdm  = off_encbf + ((size_t)64u << 20);

  unsigned* w1bf  = (unsigned*)ws;
  float*    g     = (float*)(ws + off_g);
  float*    score = (float*)(ws + off_score);
  float*    alpha = (float*)(ws + off_alpha);

  convert_w1_kernel<<<1024, 256, 0, stream>>>(W1, w1bf);
  dec_bias_kernel<<<32, 256, 0, stream>>>(dec, W1, b1, g);

  if (ws_size >= need_tdm) {
    // Preferred path: bf16 enc copy (L2-resident) + TDM-staged GEMM.
    unsigned* encbf = (unsigned*)(ws + off_encbf);
    convert_enc_kernel<<<16384, 256, 0, stream>>>(enc, (u32x4*)encbf);
    attn_score_tdm_kernel<<<dim3(16, 32), 256, 0, stream>>>(
        (const unsigned short*)encbf, w1bf, g, W2, score);
    softmax_kernel<<<32, 256, 0, stream>>>(score, alpha);
    context_bf16_kernel<<<dim3(32, 4), 128, 0, stream>>>(encbf, alpha, out);
  } else {
    attn_score_fallback_kernel<<<dim3(16, 32), 256, 0, stream>>>(
        enc, w1bf, g, W2, score);
    softmax_kernel<<<32, 256, 0, stream>>>(score, alpha);
    context_f32_kernel<<<dim3(32, 8), 128, 0, stream>>>(enc, alpha, out);
  }
}
